// MentionLinkScorer_2482491097283
// MI455X (gfx1250) — matrix operand order repaired and checked by reading, not compiled, not run on hardware
//
#include <hip/hip_runtime.h>
#include <hip/hip_bf16.h>
#include <stdint.h>

// ---------------------------------------------------------------------------
// Problem constants (from reference)
// ---------------------------------------------------------------------------
#define NM   4096      // mentions
#define NP   32768     // pairs
#define NSEG 1024      // segments
#define GD   2304      // G_DIM
#define DIN  6932      // 3*G_DIM + 20
#define HID  150       // hidden
#define KT   217       // k-tiles of 32 (216 cover 3*G_DIM=6912, +1 for phi)
#define NT   10        // n-tiles of 16 (150 -> 160 padded)
#define MBLK 64        // pairs per block (4 wmma m-tiles per wave)
#define FRAGN (KT*NT*512)             // packed W1 fragment element count
#define G16_BYTES  ((size_t)NM*GD*2)  // 18,874,368 B  f16 copy of g_i
#define W1F_BYTES  ((size_t)FRAGN*2)  //  2,222,080 B  packed W1 f16

typedef __attribute__((ext_vector_type(16))) _Float16 v16h;
typedef __attribute__((ext_vector_type(8)))  _Float16 v8h;
typedef __attribute__((ext_vector_type(4)))  _Float16 v4h;
typedef __attribute__((ext_vector_type(8)))  float    v8f;

// ---------------------------------------------------------------------------
// Kernel 1a: convert g_i (f32) -> g16 (f16), 4 elements / thread.
// ---------------------------------------------------------------------------
__global__ void mls_cvt_g(const float* __restrict__ g,
                          _Float16* __restrict__ g16) {
  int i = blockIdx.x * blockDim.x + threadIdx.x;  // < NM*GD/4
  float4 v = ((const float4*)g)[i];
  v4h h;
  h[0] = (_Float16)v.x; h[1] = (_Float16)v.y;
  h[2] = (_Float16)v.z; h[3] = (_Float16)v.w;
  ((v4h*)g16)[i] = h;
}

// ---------------------------------------------------------------------------
// Kernel 1b: pack W1 into per-lane WMMA B fragments (f16), zero-padded to
// 160 cols / 6944 rows; zero segmax/segsum scratch.
// Layout: frag[(kt*NT+nt)*512 + lane*16 + e] with
//   K = kt*32 + (e&7) + (e>>3)*16 + (lane>>4)*8,  N = nt*16 + (lane&15)
// ---------------------------------------------------------------------------
__global__ void mls_pack_w1(const float* __restrict__ W1,
                            _Float16* __restrict__ w1f,
                            int* __restrict__ segmax_bits,
                            float* __restrict__ segsum) {
  int i = blockIdx.x * blockDim.x + threadIdx.x;
  if (i < NSEG) { segmax_bits[i] = 0; segsum[i] = 0.0f; }
  if (i >= FRAGN) return;
  int e    = i & 15;
  int L    = (i >> 4) & 31;
  int tile = i >> 9;
  int nt   = tile % NT;
  int kt   = tile / NT;
  int k = kt * 32 + (e & 7) + ((e >> 3) << 4) + ((L >> 4) << 3);
  int n = nt * 16 + (L & 15);
  float v = (k < DIN && n < HID) ? W1[(size_t)k * HID + n] : 0.0f;
  w1f[i] = (_Float16)v;
}

// ---------------------------------------------------------------------------
// Kernel 2: fused pair-feature build + WMMA GEMM + MLP epilogue.
// Block = 320 threads = 10 waves (one n-tile each), M-tile = 64 pairs.
// Stage = 2 k-tiles (64x64 halves, 8 KB), double-buffered, async DMA fill.
// Three specialized section loops -> no per-iteration section branching;
// one s_wait_asynccnt + one barrier per 8 WMMAs.
// ---------------------------------------------------------------------------

#define WMMA_STEP(RB, KTG)                                                    \
  {                                                                           \
    const _Float16* bp_ = w1f + (((size_t)((KTG)*NT + wv)) << 9) + lane * 16; \
    v8h blo_ = *(const v8h*)(bp_);                                            \
    v8h bhi_ = *(const v8h*)(bp_ + 8);                                        \
    v16h bf_ = __builtin_shufflevector(blo_, bhi_,                            \
        0, 1, 2, 3, 4, 5, 6, 7, 8, 9, 10, 11, 12, 13, 14, 15);                \
    _Pragma("unroll")                                                         \
    for (int mi_ = 0; mi_ < 4; ++mi_) {                                       \
      const _Float16* rp_ = (RB) + mi_ * 512;                                 \
      v8h alo_ = *(const v8h*)(rp_);                                          \
      v8h ahi_ = *(const v8h*)(rp_ + 16);                                     \
      v16h af_ = __builtin_shufflevector(alo_, ahi_,                          \
          0, 1, 2, 3, 4, 5, 6, 7, 8, 9, 10, 11, 12, 13, 14, 15);              \
      acc[mi_] = __builtin_amdgcn_wmma_f32_16x16x32_f16(                      \
          false, af_, false, bf_, (short)0, acc[mi_], false, false);          \
    }                                                                         \
  }

#define WAIT_ASYNC() asm volatile("s_wait_asynccnt 0" ::: "memory")

// Async copy fill: stage ST (2 k-tiles) into buffer BUF from row pointer PTR.
#define FILL_ASYNC(PTR, ST, BUF)                                              \
  if (tid < 256) {                                                            \
    uint64_t ga_ = (uint64_t)(uintptr_t)((PTR) + (ST) * 64);                  \
    unsigned d_  = ldsDst0 + (unsigned)((BUF) * 8192);                        \
    asm volatile("global_load_async_to_lds_b128 %0, %1, off"                  \
                 :: "v"(d_), "v"(ga_) : "memory");                            \
    asm volatile("global_load_async_to_lds_b128 %0, %1, off"                  \
                 :: "v"(d_ + 16u), "v"(ga_ + 16) : "memory");                 \
  }

// Product fill: elementwise g16[mid]*g16[aid] via v_pk_mul_f16.
#define FILL_PROD(ST, BUF)                                                    \
  if (tid < 256) {                                                            \
    v16h a_ = *(const v16h*)(pm + (ST) * 64);                                 \
    v16h b_ = *(const v16h*)(pa + (ST) * 64);                                 \
    *(v16h*)(dstP + (BUF) * 4096) = a_ * b_;                                  \
  }

// Pipelined section: 36 stages (72 k-tiles) starting at k-tile KBASE.
#define RUN_SECTION(FILL1, KBASE)                                             \
  FILL1(0, 0);                                                                \
  WAIT_ASYNC();                                                               \
  __syncthreads();                                                            \
  for (int st = 0; st < 36; st += 2) {                                        \
    FILL1(st + 1, 1);                                                         \
    WMMA_STEP(rb0,        (KBASE) + 2 * st);                                  \
    WMMA_STEP(rb0 + 2048, (KBASE) + 2 * st + 1);                              \
    WAIT_ASYNC();                                                             \
    __syncthreads();                                                          \
    if (st < 34) FILL1(st + 2, 0);                                            \
    WMMA_STEP(rb1,        (KBASE) + 2 * st + 2);                              \
    WMMA_STEP(rb1 + 2048, (KBASE) + 2 * st + 3);                              \
    WAIT_ASYNC();                                                             \
    __syncthreads();                                                          \
  }

#define FILL_MID(ST, BUF) FILL_ASYNC(pm, ST, BUF)
#define FILL_AID(ST, BUF) FILL_ASYNC(pa, ST, BUF)

__global__ __launch_bounds__(320)
void mls_pair_gemm(const _Float16* __restrict__ g16,
                   const float* __restrict__ mention_scores,
                   const float* __restrict__ speaker_embed,
                   const _Float16* __restrict__ w1f,
                   const float* __restrict__ b1,
                   const float* __restrict__ W2,
                   const float* __restrict__ b2,
                   const int* __restrict__ mention_ids,
                   const int* __restrict__ antecedent_ids,
                   const int* __restrict__ speaker_labels,
                   const int* __restrict__ segment_ids,
                   float* __restrict__ scores,
                   int* __restrict__ segmax_bits) {
  __shared__ __align__(16) _Float16 ldsA[2][4096];     // 2 x 8 KB stage tiles
  __shared__ __align__(16) _Float16 ldsPhi[MBLK * 32]; // 4 KB phi tile
  __shared__ int   s_mid[MBLK], s_aid[MBLK];
  __shared__ float s_sij[MBLK];

  const int tid = threadIdx.x;
  const int p0  = blockIdx.x * MBLK;

  // ---- one-time init: ids, phi tile (k-invariant), partial-sum buffer ----
  if (tid < MBLK) {
    s_mid[tid] = mention_ids[p0 + tid];
    s_aid[tid] = antecedent_ids[p0 + tid];
    s_sij[tid] = 0.0f;
  }
  {
    const int fr  = tid >> 2;         // 0..63 for tid<256
    const int fcp = (tid & 3) << 3;   // 0,8,16,24
    if (tid < 256) {
      int lab = speaker_labels[p0 + fr];
      v8h v = {};
#pragma unroll
      for (int j = 0; j < 8; ++j) {
        int kk = fcp + j;
        v[j] = (kk < 20) ? (_Float16)speaker_embed[lab * 20 + kk]
                         : (_Float16)0.0f;
      }
      *(v8h*)&ldsPhi[fr * 32 + fcp] = v;
    }
  }
  __syncthreads();  // ids visible before computing fill base pointers

  // ---- per-thread fill state (loop-invariant) ----
  // Stage = 2 k-tiles stored as two consecutive 64x32 tiles.
  // row r=(tid>>1)&63, 16-half chunk co=(tid&1)*16, tile parity tpar=tid>>7.
  const int r    = (tid >> 1) & 63;
  const int co   = (tid & 1) << 4;
  const int tpar = tid >> 7;          // 0/1 for tid<256
  const _Float16* pm = g16;
  const _Float16* pa = g16;
  _Float16* dstP = &ldsPhi[0];        // placeholder for inactive threads
  unsigned ldsDst0 = 0;
  if (tid < 256) {
    pm = g16 + (size_t)s_mid[r] * GD + tpar * 32 + co;
    pa = g16 + (size_t)s_aid[r] * GD + tpar * 32 + co;
    dstP = &ldsA[0][tpar * 2048 + r * 32 + co];
    // LDS byte offset = low 32 bits of the generic (flat) address
    ldsDst0 = (unsigned)(uintptr_t)dstP;
  }

  const int lane = tid & 31;
  const int wv   = tid >> 5;          // wave id == n-tile (0..9)
  const int half = lane >> 4;
  const int nl   = lane & 15;
  const int n    = wv * 16 + nl;      // column (0..159; >=150 is zero pad)

  const _Float16* rb0 = &ldsA[0][nl * 32 + half * 8];
  const _Float16* rb1 = rb0 + 4096;   // buffer 1
  const _Float16* rbP = &ldsPhi[nl * 32 + half * 8];

  v8f acc[4] = {};

  // ---- three pipelined sections (72 k-tiles each) ----
  RUN_SECTION(FILL_MID,  0);    // rows from g16[mid]
  RUN_SECTION(FILL_AID,  72);   // rows from g16[aid]
  RUN_SECTION(FILL_PROD, 144);  // rows = g16[mid] * g16[aid]

  // ---- peeled phi tile (k-tile 216), A from ldsPhi (built at init) ----
  WMMA_STEP(rbP, 216);

  // ---- epilogue: h = relu(acc + b1); s_ij partial += h * W2[n] ----
  float b1v = 0.0f, w2v = 0.0f;
  if (n < HID) { b1v = b1[n]; w2v = W2[n]; }
#pragma unroll
  for (int mi = 0; mi < 4; ++mi) {
#pragma unroll
    for (int rr = 0; rr < 8; ++rr) {
      float h = acc[mi][rr] + b1v;
      h = h > 0.0f ? h : 0.0f;
      atomicAdd(&s_sij[mi * 16 + rr + half * 8], h * w2v);
    }
  }
  __syncthreads();

  // ---- final score + segment max (positive-only atomicMax == max(.,0)) ----
  if (tid < MBLK) {
    int p = p0 + tid;
    float sc = s_sij[tid] + b2[0] +
               mention_scores[s_mid[tid]] + mention_scores[s_aid[tid]];
    scores[p] = sc;
    if (sc > 0.0f) {
      atomicMax(&segmax_bits[segment_ids[p]], __float_as_int(sc));
    }
  }
}

// ---------------------------------------------------------------------------
// Kernel 3: e = exp(score - m); segment sums; stage e in d_out.
// ---------------------------------------------------------------------------
__global__ void mls_exp_segsum(const float* __restrict__ scores,
                               const int* __restrict__ segment_ids,
                               const int* __restrict__ segmax_bits,
                               float* __restrict__ segsum,
                               float* __restrict__ out_pairs) {
  int p = blockIdx.x * blockDim.x + threadIdx.x;
  if (p >= NP) return;
  int s = segment_ids[p];
  float m = __int_as_float(segmax_bits[s]);   // == max(seg_max, 0) >= 0
  float e = expf(scores[p] - m);
  out_pairs[p] = e;
  atomicAdd(&segsum[s], e);
}

// ---------------------------------------------------------------------------
// Kernel 4: denom = sum + exp(-m); eps probs; store denom in segsum.
// ---------------------------------------------------------------------------
__global__ void mls_eps(const int* __restrict__ segmax_bits,
                        float* __restrict__ segsum,
                        float* __restrict__ out_eps) {
  int s = blockIdx.x * blockDim.x + threadIdx.x;
  if (s >= NSEG) return;
  float m = __int_as_float(segmax_bits[s]);
  float eps = expf(-m);
  float denom = segsum[s] + eps;
  out_eps[s] = eps / denom;
  segsum[s] = denom;
}

// ---------------------------------------------------------------------------
// Kernel 5: normalize pair probs.
// ---------------------------------------------------------------------------
__global__ void mls_norm(const int* __restrict__ segment_ids,
                         const float* __restrict__ denom,
                         float* __restrict__ out_pairs) {
  int p = blockIdx.x * blockDim.x + threadIdx.x;
  if (p >= NP) return;
  out_pairs[p] = out_pairs[p] / denom[segment_ids[p]];
}

// ---------------------------------------------------------------------------
extern "C" void kernel_launch(void* const* d_in, const int* in_sizes, int n_in,
                              void* d_out, int out_size, void* d_ws, size_t ws_size,
                              hipStream_t stream) {
  const float* g_i            = (const float*)d_in[0];
  const float* mention_scores = (const float*)d_in[1];
  const float* speaker_embed  = (const float*)d_in[2];
  const float* W1             = (const float*)d_in[3];
  const float* b1             = (const float*)d_in[4];
  const float* W2             = (const float*)d_in[5];
  const float* b2             = (const float*)d_in[6];
  const int* mention_ids      = (const int*)d_in[7];
  const int* antecedent_ids   = (const int*)d_in[8];
  const int* speaker_labels   = (const int*)d_in[9];
  const int* segment_ids      = (const int*)d_in[10];

  float* out = (float*)d_out;   // [NP pair probs | NSEG eps probs]

  // Workspace layout
  char* ws = (char*)d_ws;
  _Float16* g16  = (_Float16*)ws;                              // 18,874,368 B
  _Float16* w1f  = (_Float16*)(ws + G16_BYTES);                //  2,222,080 B
  float*    scr  = (float*)(ws + G16_BYTES + W1F_BYTES);       // NP*4 B
  int*      smax = (int*)  (ws + G16_BYTES + W1F_BYTES + (size_t)NP * 4);
  float*    ssum = (float*)(ws + G16_BYTES + W1F_BYTES + (size_t)NP * 4
                               + (size_t)NSEG * 4);

  // 1a) g_i f32 -> f16
  mls_cvt_g<<<(NM * GD / 4 + 255) / 256, 256, 0, stream>>>(g_i, g16);

  // 1b) pack W1 into WMMA B fragments + zero segment scratch
  mls_pack_w1<<<(FRAGN + 255) / 256, 256, 0, stream>>>(W1, w1f, smax, ssum);

  // 2) fused pair build + WMMA GEMM + MLP epilogue + segment max
  mls_pair_gemm<<<NP / MBLK, 320, 0, stream>>>(
      g16, mention_scores, speaker_embed, w1f, b1, W2, b2,
      mention_ids, antecedent_ids, speaker_labels, segment_ids, scr, smax);

  // 3) exp + segment sums
  mls_exp_segsum<<<(NP + 255) / 256, 256, 0, stream>>>(scr, segment_ids, smax,
                                                       ssum, out);
  // 4) epsilon probs + denominators
  mls_eps<<<(NSEG + 255) / 256, 256, 0, stream>>>(smax, ssum, out + NP);

  // 5) normalize pair probs
  mls_norm<<<(NP + 255) / 256, 256, 0, stream>>>(segment_ids, ssum, out);
}
